// Int4Linear_28123445854779
// MI455X (gfx1250) — compile-verified, matrix operand order checked by reading, not run on hardware
//
#include <hip/hip_runtime.h>

typedef __attribute__((ext_vector_type(16))) _Float16 v16h;
typedef __attribute__((ext_vector_type(8)))  _Float16 v8h;
typedef __attribute__((ext_vector_type(8)))  float    v8f;
typedef __attribute__((ext_vector_type(4)))  float    f32x4;
typedef __attribute__((ext_vector_type(4)))  int      i32x4;

#define O_FEAT 11008
#define K_FEAT 4096
#define GROUP_SZ 128
#define NGROUP (K_FEAT / GROUP_SZ)
#define BM 128
#define BN 128
#define BK 64                      // 2 WMMA K-steps per staging step
#define NCH (K_FEAT / BK)          // 64 staging steps
#define NKC (K_FEAT / 32)          // 128 fragment k-chunks
#define SUB_HALVES (8 * 512)       // 8 tiles x (32 lanes x 16 halves)

// Fragment layout (ISA 7.12.2, f16 16x32 operand):
//   index((tile, kc), lane, elem) = ((tile*128 + kc) * 32 + lane) * 16 + elem
//   krel {0..7,16..23} -> lane rI (elems 0..7 / 8..15)
//   krel {8..15,24..31} -> lane rI+16 (elems 0..7 / 8..15)

__device__ __forceinline__ v8h cvt8(f32x4 a, f32x4 b) {
    v8h r;
    r[0] = (_Float16)a.x; r[1] = (_Float16)a.y;
    r[2] = (_Float16)a.z; r[3] = (_Float16)a.w;
    r[4] = (_Float16)b.x; r[5] = (_Float16)b.y;
    r[6] = (_Float16)b.z; r[7] = (_Float16)b.w;
    return r;
}

// 4 packed words -> 8 dequantized f16 (low nibble = even k, high nibble = odd k)
__device__ __forceinline__ v8h deq8(i32x4 w, float sc) {
    v8h r;
    int b;
    b = w.x; r[0] = (_Float16)((float)((b & 15) - 8) * sc);
             r[1] = (_Float16)((float)(((b >> 4) & 15) - 8) * sc);
    b = w.y; r[2] = (_Float16)((float)((b & 15) - 8) * sc);
             r[3] = (_Float16)((float)(((b >> 4) & 15) - 8) * sc);
    b = w.z; r[4] = (_Float16)((float)((b & 15) - 8) * sc);
             r[5] = (_Float16)((float)(((b >> 4) & 15) - 8) * sc);
    b = w.w; r[6] = (_Float16)((float)((b & 15) - 8) * sc);
             r[7] = (_Float16)((float)(((b >> 4) & 15) - 8) * sc);
    return r;
}

// ---------------- pre-pass: X fp32 -> f16, fragment layout ----------------
__global__ __launch_bounds__(256)
void cvt_x_frag(const float* __restrict__ X, _Float16* __restrict__ Xf) {
    const int mt = blockIdx.x;          // 16-row tile
    const int t  = threadIdx.x;
    const int rI = t & 15;
    const int q  = t >> 4;
    const float* xrow = X + ((long)mt * 16 + rI) * K_FEAT;
#pragma unroll
    for (int kk = 0; kk < 8; ++kk) {
        const int kc = q * 8 + kk;
        const float* s = xrow + kc * 32;
        f32x4 f[8];
#pragma unroll
        for (int j = 0; j < 8; ++j) f[j] = *(const f32x4*)(s + 4 * j);
        _Float16* base = Xf + ((long)mt * NKC + kc) * 512;
        *(v8h*)(base + rI * 16 + 0)        = cvt8(f[0], f[1]);  // krel 0..7
        *(v8h*)(base + rI * 16 + 8)        = cvt8(f[4], f[5]);  // krel 16..23
        *(v8h*)(base + (rI + 16) * 16 + 0) = cvt8(f[2], f[3]);  // krel 8..15
        *(v8h*)(base + (rI + 16) * 16 + 8) = cvt8(f[6], f[7]);  // krel 24..31
    }
}

// ---------------- pre-pass: W int4 -> f16 dequant, fragment layout ----------------
__global__ __launch_bounds__(256)
void deq_w_frag(const int* __restrict__ Wp, const float* __restrict__ S,
                _Float16* __restrict__ Wf) {
    const int nt = blockIdx.x;          // 16-row (out-feature) tile
    const int t  = threadIdx.x;
    const int rI = t & 15;
    const int q  = t >> 4;
    const int o  = nt * 16 + rI;
    const int*   wrow = Wp + (long)o * (K_FEAT / 2);
    const float* srow = S + (long)o * NGROUP;
#pragma unroll
    for (int kk = 0; kk < 8; ++kk) {
        const int kc = q * 8 + kk;
        const float sc = srow[kc >> 2];      // group = kc*32/128
        const int* s = wrow + kc * 16;
        i32x4 w0 = *(const i32x4*)(s + 0);
        i32x4 w1 = *(const i32x4*)(s + 4);
        i32x4 w2 = *(const i32x4*)(s + 8);
        i32x4 w3 = *(const i32x4*)(s + 12);
        _Float16* base = Wf + ((long)nt * NKC + kc) * 512;
        *(v8h*)(base + rI * 16 + 0)        = deq8(w0, sc);  // krel 0..7
        *(v8h*)(base + rI * 16 + 8)        = deq8(w2, sc);  // krel 16..23
        *(v8h*)(base + (rI + 16) * 16 + 0) = deq8(w1, sc);  // krel 8..15
        *(v8h*)(base + (rI + 16) * 16 + 8) = deq8(w3, sc);  // krel 24..31
    }
}

// ---------------- main GEMM: pure f16 WMMA, copy-only staging ----------------
// 512 threads = 16 waves; wave grid 4x4; each wave owns 2x2 16x16 tiles (acc = 32 VGPRs)
__global__ __launch_bounds__(512)
void gemm_frag_f16(const _Float16* __restrict__ Xf, const _Float16* __restrict__ Wf,
                   float* __restrict__ Y) {
    __shared__ __align__(16) _Float16 Xs[2][2][SUB_HALVES];
    __shared__ __align__(16) _Float16 Ws[2][2][SUB_HALVES];

    const int tid  = threadIdx.x;
    const int lane = tid & 31;
    const int wave = tid >> 5;
    const int wm   = wave >> 2;   // 0..3 -> m-tiles {2wm, 2wm+1}
    const int wn   = wave & 3;    // 0..3 -> n-tiles {2wn, 2wn+1}
    const int mtb  = blockIdx.y * 8;
    const int ntb  = blockIdx.x * 8;

    // staging: 16 pieces (tile x sub), 32 threads per piece, 16 halves (32B) each
    const int p      = tid >> 5;          // 0..15
    const int stSub  = p & 1;
    const int stTile = p >> 1;
    const int stOff  = (tid & 31) * 16;   // halves

    const _Float16* xsb = Xf + ((long)(mtb + stTile) * NKC + stSub) * 512 + stOff;
    const _Float16* wsb = Wf + ((long)(ntb + stTile) * NKC + stSub) * 512 + stOff;
    _Float16* xd0 = &Xs[0][stSub][stTile * 512 + stOff];
    _Float16* xd1 = &Xs[1][stSub][stTile * 512 + stOff];
    _Float16* wd0 = &Ws[0][stSub][stTile * 512 + stOff];
    _Float16* wd1 = &Ws[1][stSub][stTile * 512 + stOff];

    v8f acc[2][2] = {};
    i32x4 xr[2], wr[2];

    auto stage_load = [&](int i) {
        const i32x4* sx = (const i32x4*)(xsb + (long)i * 1024);
        const i32x4* sw = (const i32x4*)(wsb + (long)i * 1024);
        xr[0] = sx[0]; xr[1] = sx[1];
        wr[0] = sw[0]; wr[1] = sw[1];
    };
    auto stage_store = [&](int buf) {
        i32x4* dx = (i32x4*)(buf ? xd1 : xd0);
        i32x4* dw = (i32x4*)(buf ? wd1 : wd0);
        dx[0] = xr[0]; dx[1] = xr[1];
        dw[0] = wr[0]; dw[1] = wr[1];
    };
    auto compute = [&](int buf) {
#pragma unroll
        for (int s = 0; s < 2; ++s) {
            v16h a0 = *(const v16h*)&Xs[buf][s][(2 * wm + 0) * 512 + lane * 16];
            v16h a1 = *(const v16h*)&Xs[buf][s][(2 * wm + 1) * 512 + lane * 16];
            v16h b0 = *(const v16h*)&Ws[buf][s][(2 * wn + 0) * 512 + lane * 16];
            v16h b1 = *(const v16h*)&Ws[buf][s][(2 * wn + 1) * 512 + lane * 16];
            acc[0][0] = __builtin_amdgcn_wmma_f32_16x16x32_f16(
                false, a0, false, b0, (short)0, acc[0][0], false, false);
            acc[0][1] = __builtin_amdgcn_wmma_f32_16x16x32_f16(
                false, a0, false, b1, (short)0, acc[0][1], false, false);
            acc[1][0] = __builtin_amdgcn_wmma_f32_16x16x32_f16(
                false, a1, false, b0, (short)0, acc[1][0], false, false);
            acc[1][1] = __builtin_amdgcn_wmma_f32_16x16x32_f16(
                false, a1, false, b1, (short)0, acc[1][1], false, false);
        }
    };

    stage_load(0);
    stage_store(0);
    __syncthreads();

    for (int i = 0; i < NCH - 1; ++i) {
        stage_load(i + 1);              // long-latency loads first
        compute(i & 1);                 // WMMA while loads fly
        stage_store((i + 1) & 1);
        __syncthreads();
    }
    compute((NCH - 1) & 1);

    const int cm = (lane >> 4) * 8;
    const int cn = lane & 15;
#pragma unroll
    for (int at = 0; at < 2; ++at) {
#pragma unroll
        for (int bt = 0; bt < 2; ++bt) {
            const long row0 = (long)(mtb * 16 + (2 * wm + at) * 16 + cm);
            const long col  = (long)(ntb * 16 + (2 * wn + bt) * 16 + cn);
#pragma unroll
            for (int rr = 0; rr < 8; ++rr) {
                Y[(row0 + rr) * (long)O_FEAT + col] = acc[at][bt][rr];
            }
        }
    }
}

// ---------------- fallback: fused dequant GEMM (used if workspace too small) ----------------
__global__ __launch_bounds__(256)
void int4_linear_wmma_f16_db(const float* __restrict__ X,
                             const int*   __restrict__ Wp,
                             const float* __restrict__ S,
                             float*       __restrict__ Y) {
    __shared__ __align__(16) _Float16 Xs[2][2][SUB_HALVES];
    __shared__ __align__(16) _Float16 Ws[2][2][SUB_HALVES];

    const int tid  = threadIdx.x;
    const int lane = tid & 31;
    const int wave = tid >> 5;
    const int wm   = wave >> 1;
    const int wn   = wave & 1;
    const int bm = blockIdx.y * BM;
    const int bn = blockIdx.x * BN;

    const int r  = tid >> 1;
    const int h  = tid & 1;
    const int tI = r >> 4;
    const int rI = r & 15;
    const int sA0 = tI * 512 + rI * 16        + h * 8;
    const int sA1 = tI * 512 + (rI + 16) * 16 + h * 8;

    const float* xrow = X  + (long)(bm + r) * K_FEAT;
    const int*   wrow = Wp + (long)(bn + r) * (K_FEAT / 2);
    const float* srow = S  + (long)(bn + r) * NGROUP;

    v8f acc[2][4] = {};
    f32x4 xr[8];
    i32x4 wr[4];
    float sc;

    auto stage_load = [&](int k0) {
#pragma unroll
        for (int s = 0; s < 2; ++s) {
#pragma unroll
            for (int q = 0; q < 4; ++q)
                xr[s * 4 + q] = *(const f32x4*)(xrow + k0 + s * 32 + h * 16 + q * 4);
#pragma unroll
            for (int q = 0; q < 2; ++q)
                wr[s * 2 + q] = *(const i32x4*)(wrow + ((k0 + s * 32) >> 1) + h * 8 + q * 4);
        }
        sc = srow[k0 >> 7];
    };
    auto stage_store = [&](int buf) {
#pragma unroll
        for (int s = 0; s < 2; ++s) {
            *(v8h*)&Xs[buf][s][sA0] = cvt8(xr[s * 4 + 0], xr[s * 4 + 1]);
            *(v8h*)&Xs[buf][s][sA1] = cvt8(xr[s * 4 + 2], xr[s * 4 + 3]);
            *(v8h*)&Ws[buf][s][sA0] = deq8(wr[s * 2 + 0], sc);
            *(v8h*)&Ws[buf][s][sA1] = deq8(wr[s * 2 + 1], sc);
        }
    };
    auto compute = [&](int buf) {
#pragma unroll
        for (int s = 0; s < 2; ++s) {
            v16h a0 = *(const v16h*)&Xs[buf][s][(2 * wm + 0) * 512 + lane * 16];
            v16h a1 = *(const v16h*)&Xs[buf][s][(2 * wm + 1) * 512 + lane * 16];
#pragma unroll
            for (int j = 0; j < 4; ++j) {
                v16h b = *(const v16h*)&Ws[buf][s][(4 * wn + j) * 512 + lane * 16];
                acc[0][j] = __builtin_amdgcn_wmma_f32_16x16x32_f16(
                    false, a0, false, b, (short)0, acc[0][j], false, false);
                acc[1][j] = __builtin_amdgcn_wmma_f32_16x16x32_f16(
                    false, a1, false, b, (short)0, acc[1][j], false, false);
            }
        }
    };

    stage_load(0);
    stage_store(0);
    __syncthreads();
    for (int i = 0; i < NCH - 1; ++i) {
        stage_load((i + 1) * BK);
        compute(i & 1);
        stage_store((i + 1) & 1);
        __syncthreads();
    }
    compute((NCH - 1) & 1);

    const int cm = (lane >> 4) * 8;
    const int cn = lane & 15;
#pragma unroll
    for (int at = 0; at < 2; ++at) {
#pragma unroll
        for (int bt = 0; bt < 4; ++bt) {
            const long row0 = (long)(bm + (2 * wm + at) * 16 + cm);
            const long col  = (long)(bn + (4 * wn + bt) * 16 + cn);
#pragma unroll
            for (int rr = 0; rr < 8; ++rr) {
                Y[(row0 + rr) * (long)O_FEAT + col] = acc[at][bt][rr];
            }
        }
    }
}

extern "C" void kernel_launch(void* const* d_in, const int* in_sizes, int n_in,
                              void* d_out, int out_size, void* d_ws, size_t ws_size,
                              hipStream_t stream) {
    const float* x  = (const float*)d_in[0];   // [4, 2048, 4096] f32
    const int*   wp = (const int*)d_in[1];     // [11008, 2048] packed bytes in int32
    const float* sc = (const float*)d_in[2];   // [11008, 32] f32
    float*       y  = (float*)d_out;           // [4, 2048, 11008] f32

    const int Mtotal = in_sizes[0] / K_FEAT;   // 8192
    const size_t xb = (size_t)Mtotal * K_FEAT * sizeof(_Float16);   // 67 MB
    const size_t wb = (size_t)O_FEAT * K_FEAT * sizeof(_Float16);   // 90 MB

    if (ws_size >= xb + wb) {
        _Float16* Xf = (_Float16*)d_ws;
        _Float16* Wf = (_Float16*)((char*)d_ws + xb);
        cvt_x_frag<<<Mtotal / 16, 256, 0, stream>>>(x, Xf);
        deq_w_frag<<<O_FEAT / 16, 256, 0, stream>>>(wp, sc, Wf);
        dim3 grid(O_FEAT / BN, Mtotal / BM);   // (86, 64)
        gemm_frag_f16<<<grid, 512, 0, stream>>>(Xf, Wf, y);
    } else {
        dim3 grid(O_FEAT / BN, Mtotal / BM);
        int4_linear_wmma_f16_db<<<grid, 256, 0, stream>>>(x, wp, sc, y);
    }
}
